// TLGv4SelfAttention_16793322127555
// MI455X (gfx1250) — matrix operand order repaired and checked by reading, not compiled, not run on hardware
//
#include <hip/hip_runtime.h>
#include <hip/hip_bf16.h>
#include <stdint.h>

typedef unsigned int u32;
typedef unsigned short u16;
typedef __bf16 bf16_t;
typedef bf16_t v16bf __attribute__((ext_vector_type(16)));
typedef float  v8f   __attribute__((ext_vector_type(8)));
typedef int    v4i   __attribute__((ext_vector_type(4)));

#define HIDDEN   4096
#define KD       4096          // contraction dim of both GEMMs
#define NUM_HEADS 32
#define NUM_KV     8
#define HEAD_DIM 128
#define Q_PER_KV   4
#define SEQ     2048
#define QKV_OUT 6144
#define BLOCKSZ   64
#define N_LOCAL   16
#define VERT       8

// ---------------- feature probes ----------------
#if defined(__AMDGCN__) && __has_builtin(__builtin_amdgcn_global_load_async_to_lds_b128) && __has_builtin(__builtin_amdgcn_s_wait_asynccnt)
#define USE_ASYNC 1
#else
#define USE_ASYNC 0
#endif

#if USE_ASYNC
typedef __attribute__((address_space(1))) v4i* gv4i_p;   // global int4*
typedef __attribute__((address_space(3))) v4i* lv4i_p;   // LDS int4*
#endif

// ---------------- helpers ----------------
#if defined(__AMDGCN__) && __has_builtin(__builtin_amdgcn_cvt_pk_bf16_f32)
typedef bf16_t v2bf __attribute__((ext_vector_type(2)));
__device__ __forceinline__ u32 pk_bf16(float a, float b) {
  union { v2bf v; u32 u; } c;
  c.v = __builtin_amdgcn_cvt_pk_bf16_f32(a, b);
  return c.u;
}
#else
__device__ __forceinline__ u32 pk_bf16(float a, float b) {
  u32 ua = __float_as_uint(a), ub = __float_as_uint(b);
  ua += 0x7FFFu + ((ua >> 16) & 1u);
  ub += 0x7FFFu + ((ub >> 16) & 1u);
  return (ua >> 16) | (ub & 0xFFFF0000u);
}
#endif
__device__ __forceinline__ u16 f2bf(float a) {
  u32 ua = __float_as_uint(a);
  ua += 0x7FFFu + ((ua >> 16) & 1u);
  return (u16)(ua >> 16);
}
__device__ __forceinline__ float bf2f(u16 b) { return __uint_as_float((u32)b << 16); }

union FragCast { uint4 q[2]; v16bf v; };
__device__ __forceinline__ v16bf mkfrag(uint4 lo, uint4 hi) {
  FragCast f; f.q[0] = lo; f.q[1] = hi; return f.v;
}
__device__ __forceinline__ v8f wmma_bf16(v16bf a, v16bf b, v8f c) {
  return __builtin_amdgcn_wmma_f32_16x16x32_bf16(false, a, false, b, (short)0, c, false, false);
}

// 16-byte global -> LDS copy: async on CDNA5, plain copy fallback
__device__ __forceinline__ void cp16_g2l(const u16* g, u16* l) {
#if USE_ASYNC
  __builtin_amdgcn_global_load_async_to_lds_b128((gv4i_p)g, (lv4i_p)l, 0, 0);
#else
  *(uint4*)l = *(const uint4*)g;
#endif
}

// ---------------- kernel: fp32 -> bf16 convert (row-major) ----------------
__global__ __launch_bounds__(256) void k_convert(const float* __restrict__ X,
                                                 u16* __restrict__ Y) {
  int i = blockIdx.x * 256 + threadIdx.x;     // 8 elements per thread
  const float4* src = (const float4*)X + (size_t)i * 2;
  float4 a = src[0], b = src[1];
  ((uint4*)Y)[i] = make_uint4(pk_bf16(a.x, a.y), pk_bf16(a.z, a.w),
                              pk_bf16(b.x, b.y), pk_bf16(b.z, b.w));
}

// ---------------- kernel: fp32 [K][N] -> bf16 [N][K] (convert + transpose) ----------------
#define TSTR 72
__global__ __launch_bounds__(256) void k_convert_t(const float* __restrict__ W,
                                                   u16* __restrict__ Wt, int N) {
  __shared__ u16 T[64 * TSTR];
  const int tid = threadIdx.x;
  const int n0 = blockIdx.x * 64;
  const int k0 = blockIdx.y * 64;
#pragma unroll
  for (int p = 0; p < 4; p++) {
    int i = tid + p * 256;
    int r  = i >> 4;            // k row 0..63
    int c4 = (i & 15) * 4;      // n col
    float4 f = *(const float4*)(W + (size_t)(k0 + r) * N + n0 + c4);
    T[(c4+0)*TSTR + r] = f2bf(f.x);
    T[(c4+1)*TSTR + r] = f2bf(f.y);
    T[(c4+2)*TSTR + r] = f2bf(f.z);
    T[(c4+3)*TSTR + r] = f2bf(f.w);
  }
  __syncthreads();
#pragma unroll
  for (int p = 0; p < 2; p++) {
    int i = tid + p * 256;
    int r   = i >> 3;           // n row 0..63
    int seg = (i & 7) * 8;      // k segment (8 halves = 16B)
    *(uint4*)(Wt + (size_t)(n0 + r) * KD + k0 + seg) = *(const uint4*)&T[r*TSTR + seg];
  }
}

// ---------------- GEMM: bf16 A[M][KD] x bf16 Bt[N][KD], double-buffered async ----------------
#define BM 128
#define BN 128
#define BK 32
#define ASTR 40
#define BSTR 40

template<bool OUT_BF16, bool HAS_BIAS>
__global__ __launch_bounds__(256) void k_gemm(const u16* __restrict__ A,
                                              const u16* __restrict__ Bt,
                                              const float* __restrict__ bias,
                                              float* __restrict__ Cf,
                                              u16* __restrict__ Cb,
                                              int N) {
  __shared__ u16 As[2][BM * ASTR];
  __shared__ u16 Bs[2][BN * BSTR];
  const int tid  = threadIdx.x;
  const int lane = tid & 31;
  const int wid  = tid >> 5;
  const int wm   = wid >> 1;      // 0..3
  const int wn   = wid & 1;       // 0..1
  const int half = lane >> 4;
  const int lcol = lane & 15;
  const int nb = blockIdx.x * BN;
  const int mb = blockIdx.y * BM;

  v8f acc[2][4];
  const v8f vzero = {0.f,0.f,0.f,0.f,0.f,0.f,0.f,0.f};
#pragma unroll
  for (int i = 0; i < 2; i++)
#pragma unroll
    for (int j = 0; j < 4; j++) acc[i][j] = vzero;

  auto stage = [&](int kb, int buf) {
#pragma unroll
    for (int p = 0; p < 2; p++) {           // 2x(A+B) 16B transfers per thread
      int i = tid + p * 256;
      int row = i >> 2;
      int seg = (i & 3) * 8;
      cp16_g2l(A  + (size_t)(mb + row) * KD + kb + seg, &As[buf][row*ASTR + seg]);
      cp16_g2l(Bt + (size_t)(nb + row) * KD + kb + seg, &Bs[buf][row*BSTR + seg]);
    }
  };

  stage(0, 0);                              // prologue: tile 0 in flight
  for (int kb = 0; kb < KD; kb += BK) {
    const int buf = (kb >> 5) & 1;          // BK == 32
    const bool more = (kb + BK) < KD;
    if (more) stage(kb + BK, buf ^ 1);      // prefetch next tile into other buffer
#if USE_ASYNC
    if (more) __builtin_amdgcn_s_wait_asynccnt(4);  // tile i done; tile i+1 in flight
    else      __builtin_amdgcn_s_wait_asynccnt(0);
#endif
    __syncthreads();                        // all waves see the staged tile

    v16bf a[2], b[4];
#pragma unroll
    for (int mt = 0; mt < 2; mt++) {
      int row = wm*32 + mt*16 + lcol;
      int ko  = half * 8;
      uint4 lo = *(const uint4*)&As[buf][row*ASTR + ko];
      uint4 hi = *(const uint4*)&As[buf][row*ASTR + ko + 16];
      a[mt] = mkfrag(lo, hi);
    }
#pragma unroll
    for (int nt = 0; nt < 4; nt++) {
      int row = wn*64 + nt*16 + lcol;
      int ko  = half * 8;
      uint4 lo = *(const uint4*)&Bs[buf][row*BSTR + ko];
      uint4 hi = *(const uint4*)&Bs[buf][row*BSTR + ko + 16];
      b[nt] = mkfrag(lo, hi);
    }
#pragma unroll
    for (int mt = 0; mt < 2; mt++)
#pragma unroll
      for (int nt = 0; nt < 4; nt++)
        acc[mt][nt] = wmma_bf16(a[mt], b[nt], acc[mt][nt]);

    __syncthreads();                        // protect buffer reused at iter i+2
  }
#pragma unroll
  for (int mt = 0; mt < 2; mt++)
#pragma unroll
    for (int nt = 0; nt < 4; nt++) {
      int col = nb + wn*64 + nt*16 + lcol;
      float bv = HAS_BIAS ? bias[col] : 0.f;
      int rowbase = mb + wm*32 + mt*16 + half*8;
#pragma unroll
      for (int r = 0; r < 8; r++) {
        float v = acc[mt][nt][r] + bv;
        if (OUT_BF16) Cb[(size_t)(rowbase + r) * N + col] = f2bf(v);
        else          Cf[(size_t)(rowbase + r) * N + col] = v;
      }
    }
}

// ---------------- kernel: bias + RoPE + scatter to q/k/vT (bf16) ----------------
__global__ __launch_bounds__(256) void k_rope_scatter(const u16* __restrict__ qkvb,
                                                      const float* __restrict__ bias,
                                                      const int* __restrict__ pos,
                                                      u16* __restrict__ qbf,   // [t][h*128+d], pre-scaled
                                                      u16* __restrict__ kbf,   // [kv][t][d]
                                                      u16* __restrict__ vT) {  // [kv][d][t]
  const int t = blockIdx.x;
  const float p = (float)pos[t];
  const float sc = 0.08838834764831845f;   // 1/sqrt(HEAD_DIM)
  const u16* row = qkvb + (size_t)t * QKV_OUT;
  for (int c = threadIdx.x; c < QKV_OUT; c += 256) {
    int kv   = c / 768;
    int slot = (c - kv * 768) >> 7;   // 0..3=q, 4=k, 5=v
    int d    = c & 127;
    float val = bf2f(row[c]) + bias[c];
    if (slot == 5) {
      vT[((size_t)kv * HEAD_DIM + d) * SEQ + t] = f2bf(val);
    } else if (d < 64) {
      float x1 = val;
      float x2 = bf2f(row[c + 64]) + bias[c + 64];
      float jf = (float)d * (1.0f / 64.0f);
      float freq = __expf(-9.210340371976184f * jf);   // 10000^(-d/64)
      float sn, cs;
      __sincosf(p * freq, &sn, &cs);
      float o1 = x1 * cs - x2 * sn;
      float o2 = x2 * cs + x1 * sn;
      if (slot == 4) {
        u16* krow = kbf + ((size_t)kv * SEQ + t) * HEAD_DIM;
        krow[d]      = f2bf(o1);
        krow[d + 64] = f2bf(o2);
      } else {
        int h = kv * Q_PER_KV + slot;
        u16* qrow = qbf + (size_t)t * HIDDEN + h * HEAD_DIM;
        qrow[d]      = f2bf(o1 * sc);
        qrow[d + 64] = f2bf(o2 * sc);
      }
    }
  }
}

// ---------------- kernel: blocksparse flash attention ----------------
#define KSTR 136
#define VSTR 72
#define PSTR 72

__global__ __launch_bounds__(128) void k_attn(const u16* __restrict__ qbf,
                                              const u16* __restrict__ kbf,
                                              const u16* __restrict__ vT,
                                              u16* __restrict__ attn) {
  __shared__ u16 Ks[64 * KSTR];          // [token][d]
  __shared__ u16 VT[HEAD_DIM * VSTR];    // [d][token]
  __shared__ u16 Pl[4 * 16 * PSTR];      // per-wave P scratch [row][token]
  const int qb  = blockIdx.x;
  const int h   = blockIdx.y;
  const int kvh = h >> 2;
  const int tid  = threadIdx.x;
  const int lane = tid & 31;
  const int wid  = tid >> 5;
  const int half = lane >> 4;
  const int lcol = lane & 15;

  // Q fragments (already scaled bf16)
  v16bf qf[4];
  {
    const int qt = qb * 64 + wid * 16 + lcol;
    const u16* qrow = qbf + (size_t)qt * HIDDEN + h * HEAD_DIM;
#pragma unroll
    for (int ks = 0; ks < 4; ks++) {
      int d0 = ks * 32 + half * 8;
      uint4 lo = *(const uint4*)(qrow + d0);
      uint4 hi = *(const uint4*)(qrow + d0 + 16);
      qf[ks] = mkfrag(lo, hi);
    }
  }

  v8f o[8];
  float m[8], l[8];
  const v8f vzero = {0.f,0.f,0.f,0.f,0.f,0.f,0.f,0.f};
#pragma unroll
  for (int i = 0; i < 8; i++) { o[i] = vzero; m[i] = -1e30f; l[i] = 0.f; }

  const u16* kbase = kbf + (size_t)kvh * SEQ * HEAD_DIM;
  const u16* vbase = vT  + (size_t)kvh * HEAD_DIM * SEQ;

  for (int j = 0; j <= qb; j++) {
    bool allowed = (qb - j < N_LOCAL) || (((j + 1) % VERT) == 0);
    if (!allowed) continue;              // uniform over workgroup
    __syncthreads();
#pragma unroll
    for (int p = 0; p < 8; p++) {        // K block: 64x128 halves
      int i = tid + p * 128;
      int row = i >> 4;
      int seg = (i & 15) * 8;
      cp16_g2l(kbase + (size_t)(j*64 + row) * HEAD_DIM + seg, &Ks[row*KSTR + seg]);
    }
#pragma unroll
    for (int p = 0; p < 8; p++) {        // V^T block: 128x64 halves
      int i = tid + p * 128;
      int d   = i >> 3;
      int seg = (i & 7) * 8;
      cp16_g2l(vbase + (size_t)d * SEQ + j*64 + seg, &VT[d*VSTR + seg]);
    }
#if USE_ASYNC
    __builtin_amdgcn_s_wait_asynccnt(0);
#endif
    __syncthreads();

    // S = Q @ K^T
    v8f s[4];
#pragma unroll
    for (int nt = 0; nt < 4; nt++) s[nt] = vzero;
#pragma unroll
    for (int ks = 0; ks < 4; ks++) {
#pragma unroll
      for (int nt = 0; nt < 4; nt++) {
        int row = nt*16 + lcol;
        int ko  = ks*32 + half*8;
        uint4 lo = *(const uint4*)&Ks[row*KSTR + ko];
        uint4 hi = *(const uint4*)&Ks[row*KSTR + ko + 16];
        s[nt] = wmma_bf16(qf[ks], mkfrag(lo, hi), s[nt]);
      }
    }
    if (j == qb) {                       // causal mask inside diagonal block
#pragma unroll
      for (int nt = 0; nt < 4; nt++) {
        int colp = nt*16 + lcol;
#pragma unroll
        for (int r = 0; r < 8; r++) {
          int rowp = wid*16 + r + 8*half;
          if (colp > rowp) s[nt][r] = -1e30f;
        }
      }
    }
    // online softmax (row = reg + 8*half, N spread over 16 lanes)
    float rmax[8], rsum[8], scl[8];
#pragma unroll
    for (int r = 0; r < 8; r++)
      rmax[r] = fmaxf(fmaxf(s[0][r], s[1][r]), fmaxf(s[2][r], s[3][r]));
#pragma unroll
    for (int msk = 1; msk < 16; msk <<= 1)
#pragma unroll
      for (int r = 0; r < 8; r++)
        rmax[r] = fmaxf(rmax[r], __shfl_xor(rmax[r], msk, 32));
#pragma unroll
    for (int r = 0; r < 8; r++) {
      float mn = fmaxf(m[r], rmax[r]);
      scl[r] = __expf(m[r] - mn);
      m[r] = mn;
      rsum[r] = 0.f;
    }
#pragma unroll
    for (int nt = 0; nt < 4; nt++)
#pragma unroll
      for (int r = 0; r < 8; r++) {
        float pv = __expf(s[nt][r] - m[r]);
        s[nt][r] = pv;
        rsum[r] += pv;
      }
#pragma unroll
    for (int msk = 1; msk < 16; msk <<= 1)
#pragma unroll
      for (int r = 0; r < 8; r++)
        rsum[r] += __shfl_xor(rsum[r], msk, 32);
#pragma unroll
    for (int r = 0; r < 8; r++) l[r] = l[r]*scl[r] + rsum[r];
#pragma unroll
    for (int nt2 = 0; nt2 < 8; nt2++)
#pragma unroll
      for (int r = 0; r < 8; r++) o[nt2][r] *= scl[r];

    // P: D-layout -> A-layout via wave-private LDS (same-wave DS ops in order)
#pragma unroll
    for (int nt = 0; nt < 4; nt++)
#pragma unroll
      for (int r = 0; r < 8; r++)
        Pl[wid*16*PSTR + (r + 8*half)*PSTR + nt*16 + lcol] = f2bf(s[nt][r]);

    // O += P @ V
#pragma unroll
    for (int ks2 = 0; ks2 < 2; ks2++) {
      int ko = ks2*32 + half*8;
      uint4 alo = *(const uint4*)&Pl[wid*16*PSTR + lcol*PSTR + ko];
      uint4 ahi = *(const uint4*)&Pl[wid*16*PSTR + lcol*PSTR + ko + 16];
      v16bf pf = mkfrag(alo, ahi);
#pragma unroll
      for (int nt2 = 0; nt2 < 8; nt2++) {
        int drow = nt2*16 + lcol;
        int toff = ks2*32 + half*8;
        uint4 blo = *(const uint4*)&VT[drow*VSTR + toff];
        uint4 bhi = *(const uint4*)&VT[drow*VSTR + toff + 16];
        o[nt2] = wmma_bf16(pf, mkfrag(blo, bhi), o[nt2]);
      }
    }
  }
#pragma unroll
  for (int nt2 = 0; nt2 < 8; nt2++) {
    int col = h*HEAD_DIM + nt2*16 + lcol;
#pragma unroll
    for (int r = 0; r < 8; r++) {
      int row = qb*64 + wid*16 + r + 8*half;
      attn[(size_t)row * HIDDEN + col] = f2bf(o[nt2][r] / l[r]);
    }
  }
}

// ---------------- launch ----------------
extern "C" void kernel_launch(void* const* d_in, const int* in_sizes, int n_in,
                              void* d_out, int out_size, void* d_ws, size_t ws_size,
                              hipStream_t stream) {
  (void)in_sizes; (void)n_in; (void)out_size; (void)ws_size;
  const float* H    = (const float*)d_in[0];
  const int*   pos  = (const int*)d_in[1];
  const float* Wqkv = (const float*)d_in[2];
  const float* bqkv = (const float*)d_in[3];
  const float* Wd   = (const float*)d_in[4];
  const float* bd   = (const float*)d_in[5];
  float* out = (float*)d_out;

  char* ws = (char*)d_ws;
  const size_t SZ_H   = (size_t)SEQ * HIDDEN * 2;        // 16 MB
  const size_t SZ_WQ  = (size_t)QKV_OUT * KD * 2;        // 48 MB
  const size_t SZ_WD  = (size_t)HIDDEN * KD * 2;         // 32 MB
  const size_t SZ_QKV = (size_t)SEQ * QKV_OUT * 2;       // 24 MB
  const size_t SZ_Q   = (size_t)SEQ * HIDDEN * 2;        // 16 MB
  const size_t SZ_K   = (size_t)NUM_KV * SEQ * HEAD_DIM * 2;  // 4 MB
  u16* Hbf   = (u16*)(ws);
  u16* WqkvT = (u16*)(ws + SZ_H);
  u16* WdT   = (u16*)(ws + SZ_H + SZ_WQ);
  u16* qkvb  = (u16*)(ws + SZ_H + SZ_WQ + SZ_WD);
  u16* qbf   = (u16*)(ws + SZ_H + SZ_WQ + SZ_WD + SZ_QKV);
  u16* kbf   = (u16*)(ws + SZ_H + SZ_WQ + SZ_WD + SZ_QKV + SZ_Q);
  u16* vT    = (u16*)(ws + SZ_H + SZ_WQ + SZ_WD + SZ_QKV + SZ_Q + SZ_K);
  u16* attnb = (u16*)(ws + SZ_H + SZ_WQ + SZ_WD + SZ_QKV + SZ_Q + 2*SZ_K);

  k_convert<<<dim3((SEQ*HIDDEN)/(256*8)), 256, 0, stream>>>(H, Hbf);
  k_convert_t<<<dim3(QKV_OUT/64, KD/64), 256, 0, stream>>>(Wqkv, WqkvT, QKV_OUT);
  k_convert_t<<<dim3(HIDDEN/64, KD/64), 256, 0, stream>>>(Wd, WdT, HIDDEN);
  k_gemm<true,false><<<dim3(QKV_OUT/BN, SEQ/BM), 256, 0, stream>>>(
      Hbf, WqkvT, nullptr, nullptr, qkvb, QKV_OUT);
  k_rope_scatter<<<dim3(SEQ), 256, 0, stream>>>(qkvb, bqkv, pos, qbf, kbf, vT);
  k_attn<<<dim3(SEQ/BLOCKSZ, NUM_HEADS), 128, 0, stream>>>(qbf, kbf, vT, attnb);
  k_gemm<false,true><<<dim3(HIDDEN/BN, SEQ/BM), 256, 0, stream>>>(
      attnb, WdT, bd, out, nullptr, HIDDEN);
}
